// WaveNet_54056458387800
// MI455X (gfx1250) — matrix operand order repaired and checked by reading, not compiled, not run on hardware
//
#include <hip/hip_runtime.h>
#include <hip/hip_bf16.h>

// ---------------- problem constants ----------------
#define NBLK  2
#define RLAY  10
#define NL    20          // NBLK * RLAY
#define RC    32
#define DC    32
#define SC    1024
#define EC    512
#define OC    256
#define GCH   5
#define BB    2
#define LL    8192
#define RFLD  2046        // receptive field
#define TGT   (LL - RFLD) // 6146 output times per batch
#define PADL  2048        // zero left-pad of residual (>= max dilation chain)
#define OUT_T 6176        // padded stash length (>= TGT + 15)

// packed A-tile bookkeeping (each tile = 512 bf16 = 32 lanes x 16 halves)
#define LT      74                  // tiles per layer: 2+2 filter, 2+2 gate, 2 res, 64 split
#define T_LAYER 16                  // input conv uses tiles [0,16)
#define T_F1    (T_LAYER + NL * LT) // 1496
#define T_F2    (T_F1 + 32 * 32)    // 2520
#define T_TOTAL (T_F2 + 16 * 16)    // 2776

typedef __attribute__((ext_vector_type(16))) __bf16 v16bf;
typedef __attribute__((ext_vector_type(8)))  __bf16 v8bf;
typedef __attribute__((ext_vector_type(8)))  float  v8f;
typedef __attribute__((ext_vector_type(4)))  float  v4f;

static __device__ __forceinline__ v8f wmma_bf16(v16bf a, v16bf b, v8f c) {
  // D = A(16x32) * B(32x16) + C, f32 accumulate
  return __builtin_amdgcn_wmma_f32_16x16x32_bf16(false, a, false, b, (short)0, c, false, false);
}
static __device__ __forceinline__ v16bf load16(const __bf16* p) {
  return *(const v16bf*)p;
}

// Issue one 16-byte async global->LDS copy (ASYNCcnt class, CDNA5).
// ldsaddr = LDS byte address (low 32 bits of the generic shared pointer).
static __device__ __forceinline__ void async_g2l_b128(unsigned int ldsaddr,
                                                      unsigned long long gaddr) {
  asm volatile("global_load_async_to_lds_b128 %0, %1, off"
               :: "v"(ldsaddr), "v"(gaddr) : "memory");
}
static __device__ __forceinline__ void wait_asynccnt0() {
  asm volatile("s_wait_asynccnt 0x0" ::: "memory");
}

// K index inside a 32-wide K-tile for A-operand element e, lane-half h (contiguous: k = 16h + e, matches B)
static __device__ __forceinline__ int a_kmap(int e, int h) {
  return e + 16 * h;
}

// ---------------- weight packing ----------------
// Generic row-major (M,K) weight -> bf16 A tiles, tile index = mt*KT + kt.
__global__ __launch_bounds__(512) void pack_a(const float* __restrict__ W,
                                              __bf16* __restrict__ dst,
                                              int KT, int strideM, int strideK) {
  int tile = blockIdx.x;
  int mt = tile / KT, kt = tile % KT;
  int tid = threadIdx.x;              // 512 = 32 lanes x 16 elems
  int l = tid >> 4, e = tid & 15;
  int k = kt * 32 + a_kmap(e, l >> 4);
  int m = mt * 16 + (l & 15);
  dst[(size_t)tile * 512 + l * 16 + e] =
      (__bf16)W[(size_t)m * strideM + (size_t)k * strideK];
}

// Per-layer tiles: [0,2)=filter tap0, [2,4)=filter tap1, [4,6)=gate tap0,
// [6,8)=gate tap1, [8,10)=res, [10,74)=split
__global__ __launch_bounds__(512) void pack_layers(const float* __restrict__ fw,
                                                   const float* __restrict__ gw,
                                                   const float* __restrict__ rw,
                                                   const float* __restrict__ sw,
                                                   __bf16* __restrict__ wpack) {
  int l = blockIdx.y, tt = blockIdx.x;
  int tid = threadIdx.x;
  int ln = tid >> 4, e = tid & 15;
  int k = a_kmap(e, ln >> 4);         // channel index 0..31
  float val;
  if (tt < 8) {
    int grp = tt >> 2;                // 0=filter, 1=gate
    int tap = (tt >> 1) & 1;          // tap0 -> x(t-d), tap1 -> x(t)
    int m = (tt & 1) * 16 + (ln & 15);
    const float* W = grp ? gw : fw;   // (NL, 32, 32, 2)
    val = W[(size_t)l * 2048 + (size_t)m * 64 + k * 2 + tap];
  } else if (tt < 10) {
    int m = (tt - 8) * 16 + (ln & 15);
    val = rw[(size_t)l * 1024 + (size_t)m * 32 + k];      // (NL, 32, 32)
  } else {
    int m = (tt - 10) * 16 + (ln & 15);
    val = sw[(size_t)l * 32768 + (size_t)m * 32 + k];     // (NL, 1024, 32)
  }
  wpack[((size_t)(T_LAYER + l * LT + tt)) * 512 + ln * 16 + e] = (__bf16)val;
}

// zero the causal left-pad of both residual ping-pong buffers (every launch)
__global__ __launch_bounds__(256) void zero_pads(__bf16* r0, __bf16* r1) {
  size_t i = (size_t)blockIdx.x * blockDim.x + threadIdx.x;
  size_t total = (size_t)BB * PADL * RC;
  if (i < total) {
    size_t b = i / ((size_t)PADL * RC), rem = i % ((size_t)PADL * RC);
    size_t off = b * (size_t)(PADL + LL) * RC + rem;
    r0[off] = (__bf16)0.0f;
    r1[off] = (__bf16)0.0f;
  }
}

// ---------------- front 1x1 conv: 256 -> 32 ----------------
__global__ __launch_bounds__(128) void input_conv_kernel(const float* __restrict__ x,
                                                         const __bf16* __restrict__ wpack,
                                                         __bf16* __restrict__ res0) {
  const int tid = threadIdx.x;
  const int lane = tid & 31, wv = tid >> 5;
  const int n = lane & 15, hh = lane >> 4;
  const int bi = blockIdx.y;
  const int t0 = blockIdx.x * 64 + wv * 16;

  v8f zero = {};
  v8f acc[2]; acc[0] = zero; acc[1] = zero;
  for (int kt = 0; kt < 8; ++kt) {
    // gather B operand: channels (kt*32 + e + 16h) at time t0+n (x is channel-major)
    const float* src = x + ((size_t)bi * 256 + kt * 32 + 16 * hh) * (size_t)LL + (t0 + n);
    v16bf bv;
#pragma unroll
    for (int e = 0; e < 16; ++e) bv[e] = (__bf16)src[(size_t)e * LL];
#pragma unroll
    for (int mt = 0; mt < 2; ++mt)
      acc[mt] = wmma_bf16(load16(wpack + (size_t)(mt * 8 + kt) * 512 + lane * 16), bv, acc[mt]);
  }
  __bf16* rp = res0 + ((size_t)bi * (PADL + LL) + PADL + t0 + n) * (size_t)RC;
#pragma unroll
  for (int mt = 0; mt < 2; ++mt) {
    v8bf rb;
#pragma unroll
    for (int r = 0; r < 8; ++r) rb[r] = (__bf16)acc[mt][r];
    *(v8bf*)(rp + 16 * mt + 8 * hh) = rb;   // channel m = 16*mt + 8*h + r at time t
  }
}

// ---------------- gated residual layer ----------------
__global__ __launch_bounds__(128) void layer_kernel(const __bf16* __restrict__ res_in,
                                                    __bf16* __restrict__ res_out,
                                                    __bf16* __restrict__ out_ws,
                                                    const __bf16* __restrict__ wpack,
                                                    const float* __restrict__ cond,
                                                    const float* __restrict__ gcfw,
                                                    const float* __restrict__ gcsw,
                                                    int layer, int dil) {
  __shared__ alignas(16) __bf16 lds_w[10 * 512];      // 10 A-tiles (10 KB), shared by waves
  __shared__ alignas(16) __bf16 lds_out[4][16][32];   // per-wave gated output tile
  const int tid = threadIdx.x;
  const int lane = tid & 31, wv = tid >> 5;
  const int n = lane & 15, hh = lane >> 4;
  const int bi = blockIdx.y;
  const int t0 = blockIdx.x * 64 + wv * 16;
  const int t = t0 + n;

  const __bf16* rin = res_in + ((size_t)bi * (PADL + LL) + PADL) * (size_t)RC;
  __bf16* rout = res_out + ((size_t)bi * (PADL + LL) + PADL) * (size_t)RC;
  const __bf16* wl = wpack + (size_t)(T_LAYER + layer * LT) * 512;

  // stage the 10 per-layer weight tiles to LDS with async global->LDS copies
  {
    unsigned int lbase = (unsigned int)(size_t)&lds_w[0];
    unsigned long long gbase = (unsigned long long)(const void*)wl;
#pragma unroll
    for (int j = 0; j < 5; ++j) {              // 128 threads * 5 = 640 x 16B = 10 KB
      int i = tid + 128 * j;
      async_g2l_b128(lbase + (unsigned)(i * 16), gbase + (unsigned)(i * 16));
    }
    wait_asynccnt0();
  }
  __syncthreads();
  const __bf16* lw = (const __bf16*)&lds_w[0];

  // B operands: current tap and dilated tap (time-major layout -> contiguous)
  v16bf b_cur = load16(rin + (size_t)t * RC + 16 * hh);
  v16bf b_del = load16(rin + (size_t)(t - dil) * RC + 16 * hh);

  // global conditioning scalars (uniform per batch)
  float sf = 0.f, sg = 0.f;
#pragma unroll
  for (int j = 0; j < GCH; ++j) {
    float cv = cond[bi * GCH + j];
    sf += cv * gcfw[layer * GCH + j];
    sg += cv * gcsw[layer * GCH + j];
  }

  v8f zero = {};
  v8f facc[2], gacc[2];
#pragma unroll
  for (int mt = 0; mt < 2; ++mt) {
    facc[mt] = wmma_bf16(load16(lw + (size_t)(0 + mt) * 512 + lane * 16), b_del, zero);
    facc[mt] = wmma_bf16(load16(lw + (size_t)(2 + mt) * 512 + lane * 16), b_cur, facc[mt]);
    gacc[mt] = wmma_bf16(load16(lw + (size_t)(4 + mt) * 512 + lane * 16), b_del, zero);
    gacc[mt] = wmma_bf16(load16(lw + (size_t)(6 + mt) * 512 + lane * 16), b_cur, gacc[mt]);
  }

  v8bf obf[2];
#pragma unroll
  for (int mt = 0; mt < 2; ++mt) {
#pragma unroll
    for (int r = 0; r < 8; ++r) {
      float fv = tanhf(facc[mt][r] + sf);
      float gv = gacc[mt][r] + sg;
      float o = fv * (1.0f / (1.0f + expf(-gv)));
      obf[mt][r] = (__bf16)o;
    }
    *(v8bf*)&lds_out[wv][n][16 * mt + 8 * hh] = obf[mt];  // (t, ch) layout
  }
  __syncthreads();
  v16bf b_out = load16(&lds_out[wv][n][16 * hh]);         // re-formed B operand

  // residual update: res_new = Wr * out + res  (C initialized from old residual)
#pragma unroll
  for (int mt = 0; mt < 2; ++mt) {
    v8bf rv = *(const v8bf*)(rin + (size_t)t * RC + 16 * mt + 8 * hh);
    v8f c;
#pragma unroll
    for (int r = 0; r < 8; ++r) c[r] = (float)rv[r];
    v8f dres = wmma_bf16(load16(lw + (size_t)(8 + mt) * 512 + lane * 16), b_out, c);
    v8bf rb;
#pragma unroll
    for (int r = 0; r < 8; ++r) rb[r] = (__bf16)dres[r];
    *(v8bf*)(rout + (size_t)t * RC + 16 * mt + 8 * hh) = rb;
  }

  // stash gated output for skip/head (only times inside the output window)
  if (t >= RFLD) {
    __bf16* ob = out_ws + (((size_t)layer * BB + bi) * OUT_T + (size_t)(t - RFLD)) * RC;
#pragma unroll
    for (int mt = 0; mt < 2; ++mt)
      *(v8bf*)(ob + 16 * mt + 8 * hh) = obf[mt];
  }
}

// ---------------- fused skip + head (relu->f1->relu->f2) ----------------
__global__ __launch_bounds__(256) void head_kernel(const __bf16* __restrict__ out_ws,
                                                   const __bf16* __restrict__ wpack,
                                                   const float* __restrict__ f1b,
                                                   const float* __restrict__ f2b,
                                                   float* __restrict__ out) {
  __shared__ alignas(16) __bf16 lds_outs[NL][16][32]; // 20 KB: all layers' activation tiles
  __shared__ alignas(16) __bf16 lds_skip[16][SC];     // 32 KB
  __shared__ alignas(16) __bf16 lds_h[16][EC];        // 16 KB
  const int tid = threadIdx.x;
  const int lane = tid & 31, wv = tid >> 5;   // 8 waves
  const int n = lane & 15, hh = lane >> 4;
  const int bi = blockIdx.y;
  const int t0 = blockIdx.x * 16;

  // stage the 20 per-layer gated-output tiles (1 KB each) to LDS asynchronously
  {
    unsigned int lbase = (unsigned int)(size_t)&lds_outs[0][0][0];
#pragma unroll
    for (int j = 0; j < 5; ++j) {              // 256 threads * 5 = 1280 x 16B = 20 KB
      int i = tid + 256 * j;
      int l = i >> 6, sub = i & 63;
      const __bf16* gp = out_ws + (((size_t)l * BB + bi) * OUT_T + t0) * (size_t)RC;
      async_g2l_b128(lbase + (unsigned)(i * 16),
                     (unsigned long long)(const void*)gp + (unsigned)(sub * 16));
    }
    wait_asynccnt0();
  }
  __syncthreads();

  v8f zero = {};

  // phase 1: skip[1024 x 16] = sum_l Ws_l * out_l ; wave owns 128 skip channels
  v8f acc8[8];
#pragma unroll
  for (int i = 0; i < 8; ++i) acc8[i] = zero;
  for (int l = 0; l < NL; ++l) {
    v16bf bl = load16(&lds_outs[l][n][16 * hh]);
    const __bf16* wsp = wpack + (size_t)(T_LAYER + l * LT + 10) * 512;
#pragma unroll
    for (int sm = 0; sm < 8; ++sm)
      acc8[sm] = wmma_bf16(load16(wsp + (size_t)(8 * wv + sm) * 512 + lane * 16), bl, acc8[sm]);
  }
#pragma unroll
  for (int sm = 0; sm < 8; ++sm) {
    v8bf sb;
#pragma unroll
    for (int r = 0; r < 8; ++r) {
      float v = acc8[sm][r];
      sb[r] = (__bf16)(v > 0.f ? v : 0.f);   // relu
    }
    *(v8bf*)&lds_skip[n][(8 * wv + sm) * 16 + 8 * hh] = sb;
  }
  __syncthreads();

  // phase 2: h = relu(f1 * skip + b1) ; wave owns 64 of 512 h channels
  const __bf16* wf1 = wpack + (size_t)T_F1 * 512;
  v8f acc4[4];
#pragma unroll
  for (int m = 0; m < 4; ++m) {
#pragma unroll
    for (int r = 0; r < 8; ++r)
      acc4[m][r] = f1b[(4 * wv + m) * 16 + 8 * hh + r];
  }
  for (int kt = 0; kt < 32; ++kt) {
    // prefetch next iteration's weight tiles into cache while WMMAs run
    int ktn = (kt + 1) & 31;
    __builtin_prefetch(wf1 + (size_t)((4 * wv) * 32 + ktn) * 512, 0, 0);
    v16bf bk = load16(&lds_skip[n][32 * kt + 16 * hh]);
#pragma unroll
    for (int m = 0; m < 4; ++m)
      acc4[m] = wmma_bf16(load16(wf1 + (size_t)((4 * wv + m) * 32 + kt) * 512 + lane * 16),
                          bk, acc4[m]);
  }
#pragma unroll
  for (int m = 0; m < 4; ++m) {
    v8bf hb;
#pragma unroll
    for (int r = 0; r < 8; ++r) {
      float v = acc4[m][r];
      hb[r] = (__bf16)(v > 0.f ? v : 0.f);
    }
    *(v8bf*)&lds_h[n][(4 * wv + m) * 16 + 8 * hh] = hb;
  }
  __syncthreads();

  // phase 3: y = f2 * h + b2 ; wave owns 32 of 256 output channels
  const __bf16* wf2 = wpack + (size_t)T_F2 * 512;
  v8f acc2[2];
#pragma unroll
  for (int m = 0; m < 2; ++m) {
#pragma unroll
    for (int r = 0; r < 8; ++r)
      acc2[m][r] = f2b[(2 * wv + m) * 16 + 8 * hh + r];
  }
  for (int kt = 0; kt < 16; ++kt) {
    v16bf bk = load16(&lds_h[n][32 * kt + 16 * hh]);
#pragma unroll
    for (int m = 0; m < 2; ++m)
      acc2[m] = wmma_bf16(load16(wf2 + (size_t)((2 * wv + m) * 16 + kt) * 512 + lane * 16),
                          bk, acc2[m]);
  }
  int t = t0 + n;
  if (t < TGT) {
    float* op = out + ((size_t)bi * TGT + t) * OC;   // (B*tgt, 256) row-major
#pragma unroll
    for (int m = 0; m < 2; ++m) {
      v4f lo, hi;
#pragma unroll
      for (int r = 0; r < 4; ++r) { lo[r] = acc2[m][r]; hi[r] = acc2[m][4 + r]; }
      int cb = (2 * wv + m) * 16 + 8 * hh;
      *(v4f*)(op + cb) = lo;
      *(v4f*)(op + cb + 4) = hi;
    }
  }
}

// ---------------- launch ----------------
extern "C" void kernel_launch(void* const* d_in, const int* in_sizes, int n_in,
                              void* d_out, int out_size, void* d_ws, size_t ws_size,
                              hipStream_t stream) {
  const float* inputs   = (const float*)d_in[0];
  const float* cond     = (const float*)d_in[1];
  const float* input_w  = (const float*)d_in[2];
  const float* filter_w = (const float*)d_in[3];
  const float* gate_w   = (const float*)d_in[4];
  const float* res_w    = (const float*)d_in[5];
  const float* split_w  = (const float*)d_in[6];
  const float* gcf_w    = (const float*)d_in[7];
  const float* gcs_w    = (const float*)d_in[8];
  const float* f1_w     = (const float*)d_in[9];
  const float* f1_b     = (const float*)d_in[10];
  const float* f2_w     = (const float*)d_in[11];
  const float* f2_b     = (const float*)d_in[12];
  float* out = (float*)d_out;

  char* ws = (char*)d_ws;
  const size_t WPACK_BYTES = (size_t)T_TOTAL * 512 * 2;           // ~2.8 MB
  const size_t RES_BYTES   = (size_t)BB * (PADL + LL) * RC * 2;   // ~1.3 MB each
  __bf16* wpack  = (__bf16*)ws;
  __bf16* resA   = (__bf16*)(ws + WPACK_BYTES);
  __bf16* resB   = (__bf16*)(ws + WPACK_BYTES + RES_BYTES);
  __bf16* out_ws = (__bf16*)(ws + WPACK_BYTES + 2 * RES_BYTES);   // ~15.8 MB

  // 1) pack weights into WMMA A-operand layout (bf16)
  pack_a<<<dim3(16), dim3(512), 0, stream>>>(input_w, wpack, 8, 256, 1);
  pack_layers<<<dim3(LT, NL), dim3(512), 0, stream>>>(filter_w, gate_w, res_w, split_w, wpack);
  pack_a<<<dim3(32 * 32), dim3(512), 0, stream>>>(f1_w, wpack + (size_t)T_F1 * 512, 32, 1024, 1);
  pack_a<<<dim3(16 * 16), dim3(512), 0, stream>>>(f2_w, wpack + (size_t)T_F2 * 512, 16, 512, 1);

  // 2) zero causal pads of both residual buffers
  zero_pads<<<dim3((BB * PADL * RC + 255) / 256), dim3(256), 0, stream>>>(resA, resB);

  // 3) front conv 256 -> 32
  input_conv_kernel<<<dim3(LL / 64, BB), dim3(128), 0, stream>>>(inputs, wpack, resA);

  // 4) 20 gated residual layers, ping-pong residual buffers
  __bf16* bufs[2] = {resA, resB};
  for (int l = 0; l < NL; ++l) {
    int d = 1 << (l % RLAY);
    layer_kernel<<<dim3(LL / 64, BB), dim3(128), 0, stream>>>(
        bufs[l & 1], bufs[(l + 1) & 1], out_ws, wpack, cond, gcf_w, gcs_w, l, d);
  }

  // 5) fused skip accumulation + relu -> f1 -> relu -> f2 head
  head_kernel<<<dim3((TGT + 15) / 16, BB), dim3(256), 0, stream>>>(out_ws, wpack, f1_b, f2_b, out);
}
